// EncoderBlock_48498770707296
// MI455X (gfx1250) — compile-verified, matrix operand order checked by reading, not run on hardware
//
#include <hip/hip_runtime.h>
#include <math.h>

// ---------------------------------------------------------------------------
// Encoder block x2 for MI455X (gfx1250, wave32, WMMA bf16 16x16x32).
// B=2, S=4096, D=384, 6 heads, dk=64.  ~122 GFLOP vs ~45 MB working set
// (fits in 192MB L2) => compute-bound: bf16 WMMA + flash attention, with
// GLOBAL_LOAD_ASYNC_TO_LDS_B128 double-buffered tile staging (ASYNCcnt) and
// DPP16 butterfly reductions for the softmax (no LDS / DScnt stalls).
// ---------------------------------------------------------------------------

#define HEADS  6
#define DMODEL 384
#define DK     64
#define SEQ    4096
#define BATCH  2
#define NTOK   (BATCH * SEQ)

typedef __attribute__((ext_vector_type(16))) __bf16          v16bf;
typedef __attribute__((ext_vector_type(16))) unsigned short  v16u;
typedef __attribute__((ext_vector_type(8)))  float           v8f;

struct alignas(16) U128 { unsigned int w[4]; };

static __device__ __forceinline__ unsigned short f2bf(float f) {
  unsigned int u = __float_as_uint(f);
  u += 0x7FFFu + ((u >> 16) & 1u);           // round-to-nearest-even
  return (unsigned short)(u >> 16);
}

// Async global->LDS copy of one 16B chunk per lane (ASYNCcnt-tracked).
// LDS address = low 32 bits of the flat pointer (ISA 10.2: LDS aperture
// maps addr[31:0] directly to the LDS offset).
static __device__ __forceinline__ void async_cp16(const void* g, void* l) {
  unsigned lds             = (unsigned)(size_t)l;
  unsigned long long gaddr = (unsigned long long)(size_t)g;
  asm volatile("global_load_async_to_lds_b128 %0, %1, off"
               :: "v"(lds), "v"(gaddr)
               : "memory");
}

static __device__ __forceinline__ void wait_async0() {
#if __has_builtin(__builtin_amdgcn_s_wait_asynccnt)
  __builtin_amdgcn_s_wait_asynccnt(0);
#else
  asm volatile("s_wait_asynccnt 0x0" ::: "memory");
#endif
}

// ---------------------------------------------------------------------------
// DPP16 butterfly reduction over 16-lane groups (VALU-only, co-issueable;
// no DScnt serialization).  Stages: xor1, xor2 (quad_perm), quad-pair swap
// (row_half_mirror), 8-group swap (row_mirror).
// ---------------------------------------------------------------------------
#define DPP_QXOR1 0xB1   // quad_perm [1,0,3,2]
#define DPP_QXOR2 0x4E   // quad_perm [2,3,0,1]
#define DPP_RHM   0x141  // row_half_mirror
#define DPP_RM    0x140  // row_mirror

template <int CTRL>
static __device__ __forceinline__ float bfly_max(float x) {
  int t = __builtin_amdgcn_update_dpp(0, __float_as_int(x), CTRL, 0xf, 0xf, true);
  return fmaxf(x, __int_as_float(t));
}
template <int CTRL>
static __device__ __forceinline__ float bfly_add(float x) {
  int t = __builtin_amdgcn_update_dpp(0, __float_as_int(x), CTRL, 0xf, 0xf, true);
  return x + __int_as_float(t);
}
static __device__ __forceinline__ float red16_max(float x) {
  x = bfly_max<DPP_QXOR1>(x); x = bfly_max<DPP_QXOR2>(x);
  x = bfly_max<DPP_RHM>(x);   x = bfly_max<DPP_RM>(x);
  return x;
}
static __device__ __forceinline__ float red16_sum(float x) {
  x = bfly_add<DPP_QXOR1>(x); x = bfly_add<DPP_QXOR2>(x);
  x = bfly_add<DPP_RHM>(x);   x = bfly_add<DPP_RM>(x);
  return x;
}

// Build a 16-element bf16 A/B fragment from two 16-byte chunks
// (half-lane K-split layout: elems 0..7 = K[kh..kh+7], 8..15 = K[16+kh..]).
static __device__ __forceinline__ v16bf ld_frag(const unsigned short* p0,
                                                const unsigned short* p1) {
  v16u t;
  *reinterpret_cast<U128*>(&t)       = *reinterpret_cast<const U128*>(p0);
  *(reinterpret_cast<U128*>(&t) + 1) = *reinterpret_cast<const U128*>(p1);
  return __builtin_bit_cast(v16bf, t);
}

#define WMMA_BF16(a, b, c) \
  __builtin_amdgcn_wmma_f32_16x16x32_bf16(false, (a), false, (b), (short)0, (c), false, false)

// ---------------------------------------------------------------------------
// One-time f32 -> bf16 weight conversion (1.2 MB total, amortized over both
// layers; lets every GEMM tile go through the async LDS path).
// ---------------------------------------------------------------------------
__global__ void k_w2bf(const float* __restrict__ w, unsigned short* __restrict__ o) {
  int i = blockIdx.x * 256 + threadIdx.x;
  o[i] = f2bf(w[i]);
}

// ---------------------------------------------------------------------------
// Double LayerNorm (unbiased std, /(std+eps)), f32 in -> bf16 out.
// One 128-thread block per row of 384.
// ---------------------------------------------------------------------------
static __device__ __forceinline__ float block_sum(float v, float* red, int tid) {
  v = red16_sum(v);
  v += __shfl_xor(v, 16, 32);      // cross the two 16-lane rows
  int w = tid >> 5;
  __syncthreads();                 // make red[] reuse safe
  if ((tid & 31) == 0) red[w] = v;
  __syncthreads();
  return red[0] + red[1] + red[2] + red[3];
}

__global__ void k_double_ln(const float* __restrict__ x,
                            const float* __restrict__ ra, const float* __restrict__ rb,
                            const float* __restrict__ a0, const float* __restrict__ b0,
                            unsigned short* __restrict__ z) {
  __shared__ float red[4];
  const int row = blockIdx.x;
  const int tid = threadIdx.x;
  const float* xr = x + (size_t)row * DMODEL;

  float v[3], y[3];
  #pragma unroll
  for (int k = 0; k < 3; ++k) v[k] = xr[tid + k * 128];

  // --- LN 1 (ra, rb) ---
  float mean = block_sum(v[0] + v[1] + v[2], red, tid) * (1.0f / DMODEL);
  float q = 0.f;
  #pragma unroll
  for (int k = 0; k < 3; ++k) { float d = v[k] - mean; q += d * d; }
  float var = block_sum(q, red, tid) * (1.0f / (DMODEL - 1));
  float inv = 1.0f / (sqrtf(var) + 1e-6f);
  #pragma unroll
  for (int k = 0; k < 3; ++k) {
    int i = tid + k * 128;
    y[k] = ra[i] * (v[k] - mean) * inv + rb[i];
  }
  // --- LN 2 (a0, b0) ---
  float mean2 = block_sum(y[0] + y[1] + y[2], red, tid) * (1.0f / DMODEL);
  float q2 = 0.f;
  #pragma unroll
  for (int k = 0; k < 3; ++k) { float d = y[k] - mean2; q2 += d * d; }
  float var2 = block_sum(q2, red, tid) * (1.0f / (DMODEL - 1));
  float inv2 = 1.0f / (sqrtf(var2) + 1e-6f);
  #pragma unroll
  for (int k = 0; k < 3; ++k) {
    int i = tid + k * 128;
    z[(size_t)row * DMODEL + i] = f2bf(a0[i] * (y[k] - mean2) * inv2 + b0[i]);
  }
}

// ---------------------------------------------------------------------------
// Tiled WMMA GEMM: out[M,N] = A_bf16[M,K] * W_bf16[N,K]^T + bias (+ resid)
// Block tile 128x64, 8 waves (4x2), wave tile 32x32 = 2x2 wmma tiles, BK=32.
// Double-buffered LDS tiles fed by global_load_async_to_lds_b128.
// MODE 0: bf16 out.  MODE 1: f32 out + f32 residual.
// ---------------------------------------------------------------------------
template <int MODE>
__global__ void k_gemm(const unsigned short* __restrict__ A,
                       const unsigned short* __restrict__ W,
                       const float* __restrict__ bias,
                       const float* __restrict__ resid,
                       unsigned short* __restrict__ outb,
                       float* __restrict__ outf) {
  __shared__ unsigned short As[2][128][40];   // bf16, padded stride
  __shared__ unsigned short Bs[2][64][40];

  const int tid   = threadIdx.x;
  const int lane  = tid & 31;
  const int wv    = tid >> 5;
  const int waveM = wv >> 1;               // 0..3
  const int waveN = wv & 1;                // 0..1
  const int mBase = blockIdx.x * 128;
  const int nBase = blockIdx.y * 64;
  const int ln15  = lane & 15;
  const int kh    = (lane >> 4) * 8;       // half-lane K offset

  const unsigned short* Abase = A + (size_t)mBase * DMODEL;
  const unsigned short* Wbase = W + (size_t)nBase * DMODEL;

  auto issue_tiles = [&](int kb, int buf) {
    #pragma unroll
    for (int p = 0; p < 2; ++p) {          // A tile: 128x32 bf16
      int e = p * 256 + tid;
      int r = e >> 2, c = (e & 3) * 8;
      async_cp16(Abase + (size_t)r * DMODEL + kb + c, &As[buf][r][c]);
    }
    {                                      // B tile: 64x32 bf16
      int r = tid >> 2, c = (tid & 3) * 8;
      async_cp16(Wbase + (size_t)r * DMODEL + kb + c, &Bs[buf][r][c]);
    }
  };

  v8f acc[2][2] = {};

  issue_tiles(0, 0);
  wait_async0();
  __syncthreads();

  for (int kb = 0; kb < DMODEL; kb += 32) {
    const int cur = (kb >> 5) & 1;
    if (kb + 32 < DMODEL) issue_tiles(kb + 32, cur ^ 1);   // prefetch next

    v16bf af[2], bf[2];
    #pragma unroll
    for (int i = 0; i < 2; ++i) {
      const unsigned short* ar = &As[cur][waveM * 32 + i * 16 + ln15][0];
      af[i] = ld_frag(ar + kh, ar + 16 + kh);
    }
    #pragma unroll
    for (int j = 0; j < 2; ++j) {
      const unsigned short* br = &Bs[cur][waveN * 32 + j * 16 + ln15][0];
      bf[j] = ld_frag(br + kh, br + 16 + kh);
    }
    #pragma unroll
    for (int i = 0; i < 2; ++i)
      #pragma unroll
      for (int j = 0; j < 2; ++j)
        acc[i][j] = WMMA_BF16(af[i], bf[j], acc[i][j]);

    wait_async0();       // next buffer's async writes (this wave) complete
    __syncthreads();     // all waves' writes complete before consumption
  }

  // epilogue: C layout -> lane holds column ln15, rows (lane>>4)*8 + r
  #pragma unroll
  for (int i = 0; i < 2; ++i)
    #pragma unroll
    for (int j = 0; j < 2; ++j)
      #pragma unroll
      for (int r = 0; r < 8; ++r) {
        int m = mBase + waveM * 32 + i * 16 + (lane >> 4) * 8 + r;
        int n = nBase + waveN * 32 + j * 16 + ln15;
        float val = acc[i][j][r] + bias[n];
        if (MODE == 0) {
          outb[(size_t)m * DMODEL + n] = f2bf(val);
        } else {
          outf[(size_t)m * DMODEL + n] = val + resid[(size_t)m * DMODEL + n];
        }
      }
}

// ---------------------------------------------------------------------------
// Flash attention, one head & 64 q-rows per block (4 waves x 16 rows).
// Per 32-key block: S = Q*K^T (4 wmma), online softmax (DPP16 reductions),
// O += P*V (4 wmma).  K async-to-LDS; K/V double-buffered.
// ---------------------------------------------------------------------------
__global__ void k_flash(const unsigned short* __restrict__ Q,
                        const unsigned short* __restrict__ K,
                        const unsigned short* __restrict__ V,
                        unsigned short* __restrict__ O) {
  __shared__ unsigned short Ks[2][32][72];      // [key][dk]
  __shared__ unsigned short Vt[2][64][40];      // [dk][key] (transposed)
  __shared__ unsigned short Ps[4][16][40];      // per-wave P staging (C->A)

  const int tid  = threadIdx.x;
  const int lane = tid & 31;
  const int w    = tid >> 5;                 // wave id 0..3
  const int ln15 = lane & 15;
  const int kh   = (lane >> 4) * 8;
  const int qb   = blockIdx.x * 64;
  const int ho   = blockIdx.y * DK;          // head column offset
  const int rowbase = blockIdx.z * SEQ;
  const float SCALE = 0.125f;                // 1/sqrt(64)

  auto load_tiles = [&](int t, int buf) {
    #pragma unroll
    for (int p = 0; p < 2; ++p) {            // K tile [32][64], async
      int e = p * 128 + tid;
      int r = e >> 3, c = (e & 7) * 8;
      async_cp16(K + (size_t)(rowbase + t + r) * DMODEL + ho + c, &Ks[buf][r][c]);
    }
    #pragma unroll
    for (int p = 0; p < 2; ++p) {            // V tile transposed [64][32]
      int e = p * 128 + tid;
      int r = e >> 3, c = (e & 7) * 8;
      U128 tmp = *reinterpret_cast<const U128*>(
          V + (size_t)(rowbase + t + r) * DMODEL + ho + c);
      const unsigned short* ts = reinterpret_cast<const unsigned short*>(&tmp);
      #pragma unroll
      for (int j = 0; j < 8; ++j) Vt[buf][c + j][r] = ts[j];
    }
  };

  // Q fragments, resident for the whole pass (16 rows x 64 dk per wave)
  const unsigned short* qrow =
      Q + (size_t)(rowbase + qb + w * 16 + ln15) * DMODEL + ho;
  v16bf qa0 = ld_frag(qrow + kh,      qrow + 16 + kh);   // dk 0..31
  v16bf qa1 = ld_frag(qrow + 32 + kh, qrow + 48 + kh);   // dk 32..63

  v8f  acc[4] = {};
  float m_r[8], l_r[8];
  #pragma unroll
  for (int r = 0; r < 8; ++r) { m_r[r] = -INFINITY; l_r[r] = 0.f; }

  load_tiles(0, 0);
  wait_async0();
  __syncthreads();

  for (int t = 0; t < SEQ; t += 32) {
    const int cur = (t >> 5) & 1;
    if (t + 32 < SEQ) load_tiles(t + 32, cur ^ 1);       // prefetch next

    // scores: two 16x16 tiles covering 32 keys, contraction dk=64
    v8f st[2] = {};
    #pragma unroll
    for (int kt = 0; kt < 2; ++kt) {
      const unsigned short* kr = &Ks[cur][kt * 16 + ln15][0];
      v16bf b0 = ld_frag(kr + kh,      kr + 16 + kh);
      v16bf b1 = ld_frag(kr + 32 + kh, kr + 48 + kh);
      st[kt] = WMMA_BF16(qa0, b0, st[kt]);
      st[kt] = WMMA_BF16(qa1, b1, st[kt]);
    }

    // online softmax; row M = (lane>>4)*8 + r spans 16 lanes of a half-wave
    #pragma unroll
    for (int r = 0; r < 8; ++r) {
      float a  = st[0][r] * SCALE;
      float b2 = st[1][r] * SCALE;
      float mx = red16_max(fmaxf(a, b2));
      float mn   = fmaxf(m_r[r], mx);
      float corr = __expf(m_r[r] - mn);
      float p0   = __expf(a - mn);
      float p1   = __expf(b2 - mn);
      float rs   = red16_sum(p0 + p1);
      l_r[r] = l_r[r] * corr + rs;
      m_r[r] = mn;
      #pragma unroll
      for (int j = 0; j < 4; ++j) acc[j][r] *= corr;
      int row = (lane >> 4) * 8 + r;
      Ps[w][row][ln15]      = f2bf(p0);     // key tile 0
      Ps[w][row][16 + ln15] = f2bf(p1);     // key tile 1
    }

    // P*V: A = P[16q x 32key] (A layout via per-wave LDS), B = V^T columns
    v16bf pa = ld_frag(&Ps[w][ln15][kh], &Ps[w][ln15][16 + kh]);
    #pragma unroll
    for (int j = 0; j < 4; ++j) {
      const unsigned short* vr = &Vt[cur][j * 16 + ln15][0];
      v16bf bv = ld_frag(vr + kh, vr + 16 + kh);
      acc[j] = WMMA_BF16(pa, bv, acc[j]);
    }

    wait_async0();       // this wave's async K writes for next buffer done
    __syncthreads();     // whole block in sync before consuming next buffer
  }

  // normalize and write O in [B,S,D] layout (head h at cols h*64..)
  #pragma unroll
  for (int r = 0; r < 8; ++r) {
    float inv = 1.0f / l_r[r];
    int s = qb + w * 16 + (lane >> 4) * 8 + r;
    unsigned short* orow = O + (size_t)(rowbase + s) * DMODEL + ho;
    #pragma unroll
    for (int j = 0; j < 4; ++j) orow[j * 16 + ln15] = f2bf(acc[j][r] * inv);
  }
}

// ---------------------------------------------------------------------------
extern "C" void kernel_launch(void* const* d_in, const int* in_sizes, int n_in,
                              void* d_out, int out_size, void* d_ws, size_t ws_size,
                              hipStream_t stream) {
  (void)in_sizes; (void)n_in; (void)out_size; (void)ws_size;
  const float* x   = (const float*)d_in[0];
  const float* a0  = (const float*)d_in[1];
  const float* b0  = (const float*)d_in[2];
  const float* ra0 = (const float*)d_in[3];
  const float* rb0 = (const float*)d_in[4];
  const float* ra1 = (const float*)d_in[5];
  const float* rb1 = (const float*)d_in[6];
  const float* wq  = (const float*)d_in[7];
  const float* bq  = (const float*)d_in[8];
  const float* wk  = (const float*)d_in[9];
  const float* bk  = (const float*)d_in[10];
  const float* wv  = (const float*)d_in[11];
  const float* bv  = (const float*)d_in[12];
  const float* wo  = (const float*)d_in[13];
  const float* bo  = (const float*)d_in[14];

  const size_t EL = (size_t)NTOK * DMODEL;
  const size_t DD = (size_t)DMODEL * DMODEL;
  unsigned short* z   = (unsigned short*)d_ws;
  unsigned short* qb_ = z   + EL;
  unsigned short* kb_ = qb_ + EL;
  unsigned short* vb_ = kb_ + EL;
  unsigned short* ob_ = vb_ + EL;
  float*          x1  = (float*)(ob_ + EL);
  unsigned short* wqb = (unsigned short*)(x1 + EL);
  unsigned short* wkb = wqb + DD;
  unsigned short* wvb = wkb + DD;
  unsigned short* wob = wvb + DD;
  float*          out = (float*)d_out;

  dim3 gLN(NTOK);                    dim3 bLN(128);
  dim3 gGe(NTOK / 128, DMODEL / 64); dim3 bGe(256);
  dim3 gFa(SEQ / 64, HEADS, BATCH);  dim3 bFa(128);
  dim3 gW(DD / 256);                 dim3 bW(256);

  // one-time weight conversion to bf16 (reused by both layers)
  k_w2bf<<<gW, bW, 0, stream>>>(wq, wqb);
  k_w2bf<<<gW, bW, 0, stream>>>(wk, wkb);
  k_w2bf<<<gW, bW, 0, stream>>>(wv, wvb);
  k_w2bf<<<gW, bW, 0, stream>>>(wo, wob);

  // ----- layer 1 -----
  k_double_ln<<<gLN, bLN, 0, stream>>>(x, ra0, rb0, a0, b0, z);
  k_gemm<0><<<gGe, bGe, 0, stream>>>(z, wqb, bq, nullptr, qb_, nullptr);
  k_gemm<0><<<gGe, bGe, 0, stream>>>(z, wkb, bk, nullptr, kb_, nullptr);
  k_gemm<0><<<gGe, bGe, 0, stream>>>(z, wvb, bv, nullptr, vb_, nullptr);
  k_flash<<<gFa, bFa, 0, stream>>>(qb_, kb_, vb_, ob_);
  k_gemm<1><<<gGe, bGe, 0, stream>>>(ob_, wob, bo, x, nullptr, x1);

  // ----- layer 2 -----
  k_double_ln<<<gLN, bLN, 0, stream>>>(x1, ra1, rb1, a0, b0, z);
  k_gemm<0><<<gGe, bGe, 0, stream>>>(z, wqb, bq, nullptr, qb_, nullptr);
  k_gemm<0><<<gGe, bGe, 0, stream>>>(z, wkb, bk, nullptr, kb_, nullptr);
  k_gemm<0><<<gGe, bGe, 0, stream>>>(z, wvb, bv, nullptr, vb_, nullptr);
  k_flash<<<gFa, bFa, 0, stream>>>(qb_, kb_, vb_, ob_);
  k_gemm<1><<<gGe, bGe, 0, stream>>>(ob_, wob, bo, x1, nullptr, out);
}
